// ST_HMR_21534966022551
// MI455X (gfx1250) — matrix operand order repaired and checked by reading, not compiled, not run on hardware
//
#include <hip/hip_runtime.h>
#include <hip/hip_bf16.h>

// ---- problem constants ----
#define BB 32
#define TT 49
#define NBONE 24
#define HH 128
#define MROWS (BB*NBONE)          // 768
#define SZ_BTNH (BB*TT*NBONE*HH)  // 4816896
#define SZ_BNH  (BB*NBONE*HH)     // 98304
#define SZ_BTH  (BB*TT*HH)        // 200704

typedef __bf16 bf16x16 __attribute__((ext_vector_type(16)));
typedef float  f32x8   __attribute__((ext_vector_type(8)));
typedef float  f32x4   __attribute__((ext_vector_type(4)));
typedef int    i32x8   __attribute__((ext_vector_type(8)));

__device__ __forceinline__ unsigned f2bf1(float f) {
    unsigned u = __builtin_bit_cast(unsigned, f);
    u += 0x7FFFu + ((u >> 16) & 1u);          // round-to-nearest-even
    return u >> 16;
}
__device__ __forceinline__ unsigned pkbf(float lo, float hi) {
    return f2bf1(lo) | (f2bf1(hi) << 16);
}
__device__ __forceinline__ float sigmf(float x) { return 1.0f / (1.0f + __expf(-x)); }
__device__ __forceinline__ float tanhfast(float x) { return 1.0f - 2.0f / (1.0f + __expf(2.0f * x)); }
__device__ __forceinline__ f32x4 ld4(const float* p) { return *(const f32x4*)p; }

// =====================================================================
// Phase 1: fused 9-gate GEMM per timestep + LSTM-style epilogue.
// X = [h | h_{t-1} | p | h_{t+1} | s_nb | g_t | g_s]  (M=768, K=896)
// W = [U ; Wt(3 segs) ; Ws ; Zt ; Zs]                 (K=896, N=9*128)
// Double-buffered LDS pipeline: global loads for step k+1 issued before
// the WMMAs of step k so VMEM latency hides under matrix math.
// =====================================================================
__global__ __launch_bounds__(256, 4)
void st_hmr_gates_kernel(const float* __restrict__ h, const float* __restrict__ c_h,
                         const float* __restrict__ p, const float* __restrict__ g_t,
                         const float* __restrict__ c_g_t, const float* __restrict__ g_s,
                         const float* __restrict__ c_g_s,
                         const float* __restrict__ U,  const float* __restrict__ Wt,
                         const float* __restrict__ Ws, const float* __restrict__ Zt,
                         const float* __restrict__ Zs, const float* __restrict__ bias,
                         float* __restrict__ h_new, float* __restrict__ c_new_out)
{
    const int kslice = blockIdx.x;   // 0..3  (32-wide slice of H output)
    const int mblock = blockIdx.y;   // 0..11 (64 rows of B*N)
    const int t      = blockIdx.z;   // 0..48
    const int tid    = threadIdx.x;
    const int lane   = tid & 31;
    const int wave   = tid >> 5;
    const int warp_m = wave >> 1;    // 0..3 -> 16-row tile
    const int warp_k = wave & 1;     // 0..1 -> 16-col half of the 32 slice

    __shared__ unsigned short Alds[2][64][36];    // ping-pong 64x32 bf16 tiles
    __shared__ unsigned short Blds[2][32][292];   // ping-pong 32x(9*32) bf16 tiles

    f32x8 acc[9];
    #pragma unroll
    for (int g = 0; g < 9; ++g) acc[g] = f32x8{0,0,0,0,0,0,0,0};

    // A-load mapping: thread -> (row, 8-col span)
    const int lmA  = tid >> 2;
    const int c0A  = (tid & 3) * 8;
    const int rowA = mblock * 64 + lmA;
    const int bA   = rowA / NBONE, boneA = rowA % NBONE;
    const long baseBT = ((long)bA * TT + t) * NBONE;  // (b,t) row block (x NBONE)

    // B-load mapping: thread -> (row, 4-col span), looped over gates
    const int rB = tid >> 3;          // 0..31
    const int cB = (tid & 7) * 4;     // 0,4,...,28

    auto gloadA = [&](int kstep, f32x4& v0, f32x4& v1) {
        const int kk   = kstep * 32;
        const int seg  = kk >> 7;               // uniform per kstep
        const int off0 = (kk & 127) + c0A;
        v0 = f32x4{0,0,0,0}; v1 = f32x4{0,0,0,0};
        if (seg == 0) {
            const float* s = &h[(baseBT + boneA) * HH + off0];
            v0 = ld4(s); v1 = ld4(s + 4);
        } else if (seg == 1) {
            if (t > 0) { const float* s = &h[(baseBT - NBONE + boneA) * HH + off0];
                         v0 = ld4(s); v1 = ld4(s + 4); }
        } else if (seg == 2) {
            const float* s = &p[(baseBT + boneA) * HH + off0];
            v0 = ld4(s); v1 = ld4(s + 4);
        } else if (seg == 3) {
            if (t < TT-1) { const float* s = &h[(baseBT + NBONE + boneA) * HH + off0];
                            v0 = ld4(s); v1 = ld4(s + 4); }
        } else if (seg == 4) {
            if (boneA > 0) { const float* s = &h[(baseBT + boneA - 1) * HH + off0];
                             v0 = ld4(s); v1 = ld4(s + 4); }
            if (boneA < NBONE-1) { const float* s = &h[(baseBT + boneA + 1) * HH + off0];
                                   f32x4 a = ld4(s), b = ld4(s + 4);
                                   v0 = v0 + a; v1 = v1 + b; }
        } else if (seg == 5) {
            const float* s = &g_t[((long)bA * NBONE + boneA) * HH + off0];
            v0 = ld4(s); v1 = ld4(s + 4);
        } else {
            const float* s = &g_s[((long)bA * TT + t) * HH + off0];
            v0 = ld4(s); v1 = ld4(s + 4);
        }
    };
    auto gloadB = [&](int kstep, f32x4* w) {
        const int kk = kstep * 32;
        const float* Wp; int rows, lr0;
        if      (kk < 128) { Wp = U;  rows = 128; lr0 = kk;       }
        else if (kk < 512) { Wp = Wt; rows = 384; lr0 = kk - 128; }
        else if (kk < 640) { Wp = Ws; rows = 128; lr0 = kk - 512; }
        else if (kk < 768) { Wp = Zt; rows = 128; lr0 = kk - 640; }
        else               { Wp = Zs; rows = 128; lr0 = kk - 768; }
        const long gstride = (long)TT * rows * HH;
        const long rbase   = ((long)t * rows + lr0 + rB) * HH + kslice * 32 + cB;
        #pragma unroll
        for (int g = 0; g < 9; ++g) w[g] = ld4(Wp + g * gstride + rbase);
    };
    auto stA = [&](int buf, f32x4 v0, f32x4 v1) {
        unsigned* arow = (unsigned*)&Alds[buf][lmA][c0A];
        arow[0] = pkbf(v0[0], v0[1]);
        arow[1] = pkbf(v0[2], v0[3]);
        arow[2] = pkbf(v1[0], v1[1]);
        arow[3] = pkbf(v1[2], v1[3]);
    };
    auto stB = [&](int buf, const f32x4* w) {
        #pragma unroll
        for (int g = 0; g < 9; ++g) {
            unsigned* brow = (unsigned*)&Blds[buf][rB][g * 32 + cB];
            brow[0] = pkbf(w[g][0], w[g][1]);
            brow[1] = pkbf(w[g][2], w[g][3]);
        }
    };

    // ---- pipeline prologue ----
    f32x4 pa0, pa1, pw[9];
    gloadA(0, pa0, pa1);
    gloadB(0, pw);
    stA(0, pa0, pa1);
    stB(0, pw);

    const int mloc = lane & 15, halfw = lane >> 4;
    const int colbase = warp_k * 16;

    for (int kstep = 0; kstep < 28; ++kstep) {
        const int cur = kstep & 1;
        __syncthreads();
        if (kstep < 27) {            // issue next tile's VMEM early
            gloadA(kstep + 1, pa0, pa1);
            gloadB(kstep + 1, pw);
        }
        // ---- fragments + WMMA (bf16, K=32) from current buffer ----
        i32x8 apk;
        #pragma unroll
        for (int v = 0; v < 8; ++v) {
            const int k0 = ((v & 3) << 1) + (halfw << 3) + ((v >> 2) << 4);
            apk[v] = *(const int*)&Alds[cur][warp_m * 16 + mloc][k0];
        }
        const bf16x16 af = __builtin_bit_cast(bf16x16, apk);
        #pragma unroll
        for (int g = 0; g < 9; ++g) {
            i32x8 bpk;
            #pragma unroll
            for (int v = 0; v < 8; ++v) {
                const int kb = ((v & 3) << 1) + (halfw << 3) + ((v >> 2) << 4);
                bpk[v] = (int)((unsigned)Blds[cur][kb][g * 32 + colbase + mloc] | ((unsigned)Blds[cur][kb + 1][g * 32 + colbase + mloc] << 16));
            }
            acc[g] = __builtin_amdgcn_wmma_f32_16x16x32_bf16(
                false, af, false, __builtin_bit_cast(bf16x16, bpk),
                (short)0, acc[g], false, false);
        }
        if (kstep < 27) {            // stage next tile into the other buffer
            stA(1 - cur, pa0, pa1);
            stB(1 - cur, pw);
        }
    }

    // ---- epilogue: sigmoid/tanh gates, build c_new / h_new in-register ----
    const int nloc = lane & 15, halfe = lane >> 4;
    const int k = kslice * 32 + warp_k * 16 + nloc;
    float bsv[9];
    #pragma unroll
    for (int g = 0; g < 9; ++g) bsv[g] = bias[((long)g * TT + t) * HH + k];

    #pragma unroll
    for (int r = 0; r < 8; ++r) {
        const int m   = r + halfe * 8;
        const int row = mblock * 64 + warp_m * 16 + m;
        const int b   = row / NBONE, bone = row % NBONE;
        const long base = (((long)b * TT + t) * NBONE + bone) * HH + k;
        const float i_g  = sigmf(acc[0][r] + bsv[0]);
        const float lt_g = sigmf(acc[1][r] + bsv[1]);
        const float ft_g = sigmf(acc[2][r] + bsv[2]);
        const float rt_g = sigmf(acc[3][r] + bsv[3]);
        const float sg_g = sigmf(acc[4][r] + bsv[4]);
        const float gtg  = sigmf(acc[5][r] + bsv[5]);
        const float gsg  = sigmf(acc[6][r] + bsv[6]);
        const float o_g  = sigmf(acc[7][r] + bsv[7]);
        const float ctil = tanhfast(acc[8][r] + bsv[8]);
        const float chm = (t > 0)          ? c_h[base - NBONE * HH] : 0.f;
        const float ch  =                    c_h[base];
        const float chp = (t < TT - 1)     ? c_h[base + NBONE * HH] : 0.f;
        const float cl  = (bone > 0)         ? c_h[base - HH] : 0.f;
        const float cr  = (bone < NBONE - 1) ? c_h[base + HH] : 0.f;
        const float cgt = c_g_t[((long)b * NBONE + bone) * HH + k];
        const float cgs = c_g_s[((long)b * TT + t) * HH + k];
        const float cn = i_g * ctil + lt_g * chm + ft_g * ch + rt_g * chp
                       + sg_g * (cl + cr) + gtg * cgt + gsg * cgs;
        c_new_out[base] = cn;
        h_new[base]     = o_g * tanhfast(cn);
    }
}

// =====================================================================
// Phase 2: fused 6-gate global-state GEMM + atomic mean accumulation.
// X2 = [h_new | g_t | g_s]  (K=384);  W2 = [Wg ; Zg(t-gates) ; Zg(s-gates)]
// =====================================================================
__global__ __launch_bounds__(256, 4)
void st_hmr_global_kernel(const float* __restrict__ h_new, const float* __restrict__ c_new,
                          const float* __restrict__ g_t, const float* __restrict__ g_s,
                          const float* __restrict__ Wg, const float* __restrict__ Zg,
                          const float* __restrict__ bg,
                          float* __restrict__ acc_ft, float* __restrict__ acc_ggt,
                          float* __restrict__ acc_ot, float* __restrict__ acc_fs,
                          float* __restrict__ acc_ggs, float* __restrict__ acc_os)
{
    const int kslice = blockIdx.x;   // 0..3
    const int mblock = blockIdx.y;   // 0..11
    const int t      = blockIdx.z;   // 0..48
    const int tid    = threadIdx.x;
    const int lane   = tid & 31;
    const int wave   = tid >> 5;
    const int warp_m = wave >> 1, warp_k = wave & 1;

    __shared__ unsigned short Alds[2][64][36];
    __shared__ unsigned short Blds[2][32][196];   // ping-pong 32 x (6*32)

    f32x8 acc[6];
    #pragma unroll
    for (int g = 0; g < 6; ++g) acc[g] = f32x8{0,0,0,0,0,0,0,0};

    const int lmA  = tid >> 2;
    const int c0A  = (tid & 3) * 8;
    const int rowA = mblock * 64 + lmA;
    const int bA   = rowA / NBONE, boneA = rowA % NBONE;
    const long baseBT = ((long)bA * TT + t) * NBONE;

    const int rB = tid >> 3;
    const int cB = (tid & 7) * 4;

    auto gloadA = [&](int kstep, f32x4& v0, f32x4& v1) {
        const int kk   = kstep * 32;
        const int seg  = kk >> 7;
        const int off0 = (kk & 127) + c0A;
        const float* s;
        if      (seg == 0) s = &h_new[(baseBT + boneA) * HH + off0];
        else if (seg == 1) s = &g_t[((long)bA * NBONE + boneA) * HH + off0];
        else               s = &g_s[((long)bA * TT + t) * HH + off0];
        v0 = ld4(s); v1 = ld4(s + 4);
    };
    auto gloadB = [&](int kstep, f32x4* w) {
        const int kk = kstep * 32;
        const int seg = kk >> 7;
        const float* Wp; int lr0;
        if      (seg == 0) { Wp = Wg; lr0 = kk;       }
        else if (seg == 1) { Wp = Zg; lr0 = kk - 128; }
        else               { Wp = Zg; lr0 = kk - 256; }
        const long gstride = (long)TT * 128 * HH;
        const long rbase   = ((long)t * 128 + lr0 + rB) * HH + kslice * 32 + cB;
        #pragma unroll
        for (int g = 0; g < 6; ++g) {
            const bool active = (seg == 0) || (seg == 1 && g < 3) || (seg == 2 && g >= 3);
            w[g] = active ? ld4(Wp + g * gstride + rbase) : f32x4{0,0,0,0};
        }
    };
    auto stA = [&](int buf, f32x4 v0, f32x4 v1) {
        unsigned* arow = (unsigned*)&Alds[buf][lmA][c0A];
        arow[0] = pkbf(v0[0], v0[1]);
        arow[1] = pkbf(v0[2], v0[3]);
        arow[2] = pkbf(v1[0], v1[1]);
        arow[3] = pkbf(v1[2], v1[3]);
    };
    auto stB = [&](int buf, const f32x4* w) {
        #pragma unroll
        for (int g = 0; g < 6; ++g) {
            unsigned* brow = (unsigned*)&Blds[buf][rB][g * 32 + cB];
            brow[0] = pkbf(w[g][0], w[g][1]);
            brow[1] = pkbf(w[g][2], w[g][3]);
        }
    };

    // ---- pipeline prologue ----
    f32x4 pa0, pa1, pw[6];
    gloadA(0, pa0, pa1);
    gloadB(0, pw);
    stA(0, pa0, pa1);
    stB(0, pw);

    const int mloc = lane & 15, halfw = lane >> 4;
    const int colbase = warp_k * 16;

    for (int kstep = 0; kstep < 12; ++kstep) {
        const int cur = kstep & 1;
        __syncthreads();
        if (kstep < 11) {
            gloadA(kstep + 1, pa0, pa1);
            gloadB(kstep + 1, pw);
        }
        i32x8 apk;
        #pragma unroll
        for (int v = 0; v < 8; ++v) {
            const int k0 = ((v & 3) << 1) + (halfw << 3) + ((v >> 2) << 4);
            apk[v] = *(const int*)&Alds[cur][warp_m * 16 + mloc][k0];
        }
        const bf16x16 af = __builtin_bit_cast(bf16x16, apk);
        #pragma unroll
        for (int g = 0; g < 6; ++g) {
            i32x8 bpk;
            #pragma unroll
            for (int v = 0; v < 8; ++v) {
                const int kb = ((v & 3) << 1) + (halfw << 3) + ((v >> 2) << 4);
                bpk[v] = (int)((unsigned)Blds[cur][kb][g * 32 + colbase + mloc] | ((unsigned)Blds[cur][kb + 1][g * 32 + colbase + mloc] << 16));
            }
            acc[g] = __builtin_amdgcn_wmma_f32_16x16x32_bf16(
                false, af, false, __builtin_bit_cast(bf16x16, bpk),
                (short)0, acc[g], false, false);
        }
        if (kstep < 11) {
            stA(1 - cur, pa0, pa1);
            stB(1 - cur, pw);
        }
    }

    // ---- epilogue: sigmoids, accumulate t-means and n-means via atomics ----
    const int nloc = lane & 15, halfe = lane >> 4;
    const int k = kslice * 32 + warp_k * 16 + nloc;
    float bsv[6];
    #pragma unroll
    for (int g = 0; g < 6; ++g) bsv[g] = bg[((long)g * TT + t) * HH + k];

    #pragma unroll
    for (int r = 0; r < 8; ++r) {
        const int m   = r + halfe * 8;
        const int row = mblock * 64 + warp_m * 16 + m;
        const int b   = row / NBONE, bone = row % NBONE;
        const long base = (((long)b * TT + t) * NBONE + bone) * HH + k;
        const float fT = sigmf(acc[0][r] + bsv[0]);
        const float gT = sigmf(acc[1][r] + bsv[1]);
        const float oT = sigmf(acc[2][r] + bsv[2]);
        const float fS = sigmf(acc[3][r] + bsv[3]);
        const float gS = sigmf(acc[4][r] + bsv[4]);
        const float oS = sigmf(acc[5][r] + bsv[5]);
        const float cn = c_new[base];
        const long it = ((long)b * NBONE + bone) * HH + k;   // [B,N,H] accumulators
        const long is = ((long)b * TT + t) * HH + k;         // [B,T,H] accumulators
        __hip_atomic_fetch_add(&acc_ft[it],  fT * cn, __ATOMIC_RELAXED, __HIP_MEMORY_SCOPE_AGENT);
        __hip_atomic_fetch_add(&acc_ggt[it], gT,      __ATOMIC_RELAXED, __HIP_MEMORY_SCOPE_AGENT);
        __hip_atomic_fetch_add(&acc_ot[it],  oT,      __ATOMIC_RELAXED, __HIP_MEMORY_SCOPE_AGENT);
        __hip_atomic_fetch_add(&acc_fs[is],  fS * cn, __ATOMIC_RELAXED, __HIP_MEMORY_SCOPE_AGENT);
        __hip_atomic_fetch_add(&acc_ggs[is], gS,      __ATOMIC_RELAXED, __HIP_MEMORY_SCOPE_AGENT);
        __hip_atomic_fetch_add(&acc_os[is],  oS,      __ATOMIC_RELAXED, __HIP_MEMORY_SCOPE_AGENT);
    }
}

// ---- small helpers ----
__global__ void zero_kernel(float* p, int n) {
    int i = blockIdx.x * 256 + threadIdx.x;
    if (i < n) p[i] = 0.f;
}
__global__ void fin_t_kernel(const float* acc_ft, const float* acc_ggt, const float* acc_ot,
                             const float* c_g_t, float* g_t_new, float* c_g_t_new) {
    int i = blockIdx.x * 256 + threadIdx.x;
    if (i >= SZ_BNH) return;
    const float inv = 1.0f / (float)TT;
    float c = acc_ft[i] * inv + (acc_ggt[i] * inv) * c_g_t[i];
    c_g_t_new[i] = c;
    g_t_new[i]   = (acc_ot[i] * inv) * tanhfast(c);
}
__global__ void fin_s_kernel(const float* acc_fs, const float* acc_ggs, const float* acc_os,
                             const float* c_g_s, float* g_s_new, float* c_g_s_new) {
    int i = blockIdx.x * 256 + threadIdx.x;
    if (i >= SZ_BTH) return;
    const float inv = 1.0f / (float)NBONE;
    float c = acc_fs[i] * inv + (acc_ggs[i] * inv) * c_g_s[i];
    c_g_s_new[i] = c;
    g_s_new[i]   = (acc_os[i] * inv) * tanhfast(c);
}

extern "C" void kernel_launch(void* const* d_in, const int* in_sizes, int n_in,
                              void* d_out, int out_size, void* d_ws, size_t ws_size,
                              hipStream_t stream) {
    const float* h     = (const float*)d_in[0];
    const float* c_h   = (const float*)d_in[1];
    const float* p     = (const float*)d_in[2];
    const float* g_t   = (const float*)d_in[3];
    const float* c_g_t = (const float*)d_in[4];
    const float* g_s   = (const float*)d_in[5];
    const float* c_g_s = (const float*)d_in[6];
    const float* U     = (const float*)d_in[7];
    const float* Wt    = (const float*)d_in[8];
    const float* Ws    = (const float*)d_in[9];
    const float* Zt    = (const float*)d_in[10];
    const float* Zs    = (const float*)d_in[11];
    const float* bias  = (const float*)d_in[12];
    const float* Wg    = (const float*)d_in[13];
    const float* Zg    = (const float*)d_in[14];
    const float* bg    = (const float*)d_in[15];

    float* out    = (float*)d_out;
    float* h_new  = out;                                  // [B,T,N,H]
    float* c_new  = out + SZ_BTNH;                        // [B,T,N,H]
    float* gtn    = out + 2 * (long)SZ_BTNH;              // [B,N,H]
    float* cgtn   = gtn + SZ_BNH;                         // [B,N,H]
    float* gsn    = cgtn + SZ_BNH;                        // [B,T,H]
    float* cgsn   = gsn + SZ_BTH;                         // [B,T,H]

    float* ws      = (float*)d_ws;                        // 3.6 MB of accumulators
    float* acc_ft  = ws;
    float* acc_ggt = acc_ft  + SZ_BNH;
    float* acc_ot  = acc_ggt + SZ_BNH;
    float* acc_fs  = acc_ot  + SZ_BNH;
    float* acc_ggs = acc_fs  + SZ_BTH;
    float* acc_os  = acc_ggs + SZ_BTH;
    const int acctot = 3 * SZ_BNH + 3 * SZ_BTH;

    zero_kernel<<<(acctot + 255) / 256, 256, 0, stream>>>(ws, acctot);

    dim3 grid(4, 12, TT);   // kslice x mblock x t
    st_hmr_gates_kernel<<<grid, 256, 0, stream>>>(h, c_h, p, g_t, c_g_t, g_s, c_g_s,
                                                  U, Wt, Ws, Zt, Zs, bias, h_new, c_new);
    st_hmr_global_kernel<<<grid, 256, 0, stream>>>(h_new, c_new, g_t, g_s, Wg, Zg, bg,
                                                   acc_ft, acc_ggt, acc_ot,
                                                   acc_fs, acc_ggs, acc_os);
    fin_t_kernel<<<(SZ_BNH + 255) / 256, 256, 0, stream>>>(acc_ft, acc_ggt, acc_ot,
                                                           c_g_t, gtn, cgtn);
    fin_s_kernel<<<(SZ_BTH + 255) / 256, 256, 0, stream>>>(acc_fs, acc_ggs, acc_os,
                                                           c_g_s, gsn, cgsn);
}
